// VQEmbedding_13786845020684
// MI455X (gfx1250) — matrix operand order repaired and checked by reading, not compiled, not run on hardware
//
#include <hip/hip_runtime.h>

typedef __attribute__((ext_vector_type(16))) _Float16 v16h;
typedef __attribute__((ext_vector_type(8)))  float    v8f;

#define KCODES 512
#define DDIM   256
#define NVEC   131072
#define WAVES_PER_WG 8
#define NWG    256
#define CB_STRIDE 264   // padded f16 row stride (528B) -> conflict-free ds_load_b128

union V16U { uint4 q[2]; v16h v; };

__device__ __forceinline__ void ld_b(V16U& b, const _Float16* p) {
  b.q[0] = *reinterpret_cast<const uint4*>(p);
  b.q[1] = *reinterpret_cast<const uint4*>(p + 16);
}

__device__ __forceinline__ unsigned int pack_f16x2(float a, float b) {
  union { _Float16 h; unsigned short u; } x, y;
  x.h = (_Float16)a; y.h = (_Float16)b;
  return (unsigned int)x.u | ((unsigned int)y.u << 16);
}

#define WMMA_F16(A, B, C) \
  __builtin_amdgcn_wmma_f32_16x16x32_f16(false, (A), false, (B), (short)0, (C), false, false)

extern "C" __global__ void __launch_bounds__(WAVES_PER_WG * 32)
vq_wmma_kernel(const float* __restrict__ z_e,
               const float* __restrict__ cb,
               float* __restrict__ out) {
  extern __shared__ unsigned char smem_raw[];
  _Float16* cbh = reinterpret_cast<_Float16*>(smem_raw);                 // [512][264] f16
  float* nhnorm = reinterpret_cast<float*>(smem_raw + KCODES * CB_STRIDE * 2); // [512] f32, -0.5*||e||^2

  const int tid  = threadIdx.x;
  const int lane = tid & 31;
  const int wave = tid >> 5;

  // ---- stage 1: codebook f32 -> f16 into padded LDS (once per WG) ----
  for (int i = tid; i < KCODES * DDIM / 2; i += WAVES_PER_WG * 32) {
    const int k  = i >> 7;          // 128 float2 per code row
    const int dp = i & 127;
    float2 f = reinterpret_cast<const float2*>(cb)[i];
    *reinterpret_cast<unsigned int*>(cbh + k * CB_STRIDE + dp * 2) = pack_f16x2(f.x, f.y);
  }
  // negated half squared norms in f32 (bias applied at compare time)
  for (int k = tid; k < KCODES; k += WAVES_PER_WG * 32) {
    const float4* row = reinterpret_cast<const float4*>(cb + (size_t)k * DDIM);
    float s = 0.f;
#pragma unroll 8
    for (int j = 0; j < DDIM / 4; ++j) {
      float4 v = row[j];
      s += v.x * v.x + v.y * v.y + v.z * v.z + v.w * v.w;
    }
    nhnorm[k] = -0.5f * s;
  }
  __syncthreads();

  const int lm  = lane & 15;
  const int hi  = lane >> 4;        // half-wave select
  const int kbA = hi * 8;           // A fragment K base within a 32-chunk
  const int kbB = hi * 8;           // B fragment K base within a 32-chunk

  const int wid    = blockIdx.x * WAVES_PER_WG + wave;   // 0..2047
  const int nwaves = NWG * WAVES_PER_WG;                 // 2048
  const int iters  = (NVEC / 16) / nwaves;               // 4 (exact)

  const _Float16* bp0 = cbh + lm * CB_STRIDE + kbB;      // k-tile 0 base for this lane

  for (int it = 0; it < iters; ++it) {
    const int tile = wid + it * nwaves;
    const int row0 = tile * 16;

    // ---- load + convert A fragments: 16 rows x 256 d (f32 -> f16, ISA layout) ----
    v16h afrag[8];
    const float* arow = z_e + (size_t)(row0 + lm) * DDIM;
#pragma unroll
    for (int c = 0; c < 8; ++c) {
      const int d0 = c * 32 + kbA;
      float4 f0 = *reinterpret_cast<const float4*>(arow + d0);
      float4 f1 = *reinterpret_cast<const float4*>(arow + d0 + 4);
      float4 f2 = *reinterpret_cast<const float4*>(arow + d0 + 16);
      float4 f3 = *reinterpret_cast<const float4*>(arow + d0 + 20);
      v16h a;
      a[0]  = (_Float16)f0.x; a[1]  = (_Float16)f0.y; a[2]  = (_Float16)f0.z; a[3]  = (_Float16)f0.w;
      a[4]  = (_Float16)f1.x; a[5]  = (_Float16)f1.y; a[6]  = (_Float16)f1.z; a[7]  = (_Float16)f1.w;
      a[8]  = (_Float16)f2.x; a[9]  = (_Float16)f2.y; a[10] = (_Float16)f2.z; a[11] = (_Float16)f2.w;
      a[12] = (_Float16)f3.x; a[13] = (_Float16)f3.y; a[14] = (_Float16)f3.z; a[15] = (_Float16)f3.w;
      afrag[c] = a;
    }

    // prefetch next tile's A rows into cache while we sweep the codebook
    if (it + 1 < iters) {
      const float* pn = z_e + (size_t)((wid + (it + 1) * nwaves) * 16 + lm) * DDIM + hi * 128;
#pragma unroll
      for (int j = 0; j < 4; ++j)
        __builtin_prefetch(pn + j * 32, 0, 1);
    }

    float best[8]; int bidx[8];
#pragma unroll
    for (int r = 0; r < 8; ++r) { best[r] = -3.0e38f; bidx[r] = 0; }

    // ---- sweep all 512 codes in 16-code tiles ----
    // 4 rotating B buffers, software-pipelined across chunks AND across k-tiles:
    // each WMMA runs with 6 younger ds_loads outstanding.
    V16U b0, b1, b2, b3;
    ld_b(b0, bp0);
    ld_b(b1, bp0 + 32);
    ld_b(b2, bp0 + 64);
    ld_b(b3, bp0 + 96);

    for (int t = 0; t < KCODES / 16; ++t) {
      const int n = t * 16 + lm;
      const float nhn = nhnorm[n];       // LDS load; only needed at compare time
      v8f acc0, acc1;
#pragma unroll
      for (int r = 0; r < 8; ++r) { acc0[r] = 0.0f; acc1[r] = 0.0f; }

      const _Float16* bp  = bp0 + t * (16 * CB_STRIDE);
      const _Float16* bpn = (t == KCODES / 16 - 1) ? bp0 : bp + 16 * CB_STRIDE;

      acc0 = WMMA_F16(afrag[0], b0.v, acc0);  ld_b(b0, bp + 4 * 32);
      acc1 = WMMA_F16(afrag[1], b1.v, acc1);  ld_b(b1, bp + 5 * 32);
      acc0 = WMMA_F16(afrag[2], b2.v, acc0);  ld_b(b2, bp + 6 * 32);
      acc1 = WMMA_F16(afrag[3], b3.v, acc1);  ld_b(b3, bp + 7 * 32);
      acc0 = WMMA_F16(afrag[4], b0.v, acc0);  ld_b(b0, bpn);            // next k-tile chunk 0
      acc1 = WMMA_F16(afrag[5], b1.v, acc1);  ld_b(b1, bpn + 32);
      acc0 = WMMA_F16(afrag[6], b2.v, acc0);  ld_b(b2, bpn + 64);
      acc1 = WMMA_F16(afrag[7], b3.v, acc1);  ld_b(b3, bpn + 96);

#pragma unroll
      for (int r = 0; r < 8; ++r) {
        const float sc = acc0[r] + acc1[r] + nhn;        // biased score
        if (sc > best[r]) { best[r] = sc; bidx[r] = n; } // strict > keeps lowest idx
      }
    }

    // ---- argmax butterfly across the 16 lanes of each half-wave ----
#pragma unroll
    for (int off = 1; off < 16; off <<= 1) {
#pragma unroll
      for (int r = 0; r < 8; ++r) {
        float ov = __shfl_xor(best[r], off, 32);
        int   oi = __shfl_xor(bidx[r], off, 32);
        if (ov > best[r] || (ov == best[r] && oi < bidx[r])) { best[r] = ov; bidx[r] = oi; }
      }
    }

    // ---- gather winning codes (f32) and write both tuple outputs ----
    float* out0 = out;
    float* out1 = out + (size_t)NVEC * DDIM;
#pragma unroll
    for (int r = 0; r < 8; ++r) {
      const int idxLo = __builtin_amdgcn_readlane(bidx[r], 0);   // rows 0..7
      const int idxHi = __builtin_amdgcn_readlane(bidx[r], 16);  // rows 8..15
      {
        const float4* src = reinterpret_cast<const float4*>(cb + (size_t)idxLo * DDIM);
        float4* d0 = reinterpret_cast<float4*>(out0 + (size_t)(row0 + r) * DDIM);
        float4* d1 = reinterpret_cast<float4*>(out1 + (size_t)(row0 + r) * DDIM);
#pragma unroll
        for (int j = 0; j < 2; ++j) {
          float4 v = src[lane + j * 32];
          d0[lane + j * 32] = v;
          d1[lane + j * 32] = v;
        }
      }
      {
        const float4* src = reinterpret_cast<const float4*>(cb + (size_t)idxHi * DDIM);
        float4* d0 = reinterpret_cast<float4*>(out0 + (size_t)(row0 + 8 + r) * DDIM);
        float4* d1 = reinterpret_cast<float4*>(out1 + (size_t)(row0 + 8 + r) * DDIM);
#pragma unroll
        for (int j = 0; j < 2; ++j) {
          float4 v = src[lane + j * 32];
          d0[lane + j * 32] = v;
          d1[lane + j * 32] = v;
        }
      }
    }
  }
}

extern "C" void kernel_launch(void* const* d_in, const int* in_sizes, int n_in,
                              void* d_out, int out_size, void* d_ws, size_t ws_size,
                              hipStream_t stream) {
  (void)in_sizes; (void)n_in; (void)out_size; (void)d_ws; (void)ws_size;
  const float* z_e = (const float*)d_in[0];
  const float* cb  = (const float*)d_in[1];
  float* out = (float*)d_out;

  const size_t lds_bytes = (size_t)KCODES * CB_STRIDE * 2 + KCODES * sizeof(float); // 272,384 B
  (void)hipFuncSetAttribute(reinterpret_cast<const void*>(vq_wmma_kernel),
                            hipFuncAttributeMaxDynamicSharedMemorySize,
                            (int)lds_bytes);
  vq_wmma_kernel<<<NWG, WAVES_PER_WG * 32, lds_bytes, stream>>>(z_e, cb, out);
}
